// DIN_51926154608651
// MI455X (gfx1250) — matrix operand order repaired and checked by reading, not compiled, not run on hardware
//
#include <hip/hip_runtime.h>
#include <hip/hip_bf16.h>
#include <math.h>

typedef __attribute__((ext_vector_type(16))) _Float16 v16h;
typedef __attribute__((ext_vector_type(8)))  _Float16 v8h;
typedef __attribute__((ext_vector_type(8)))  float    v8f;

#define BN_EPS 1e-5f

// ---------------------------------------------------------------------------
// Zero-init workspace accumulators
// ---------------------------------------------------------------------------
__global__ void zero_kernel(float* p, int n) {
    int i = blockIdx.x * blockDim.x + threadIdx.x;
    if (i < n) p[i] = 0.0f;
}

// ---------------------------------------------------------------------------
// Pre-swizzle weights W[KxN] f32 -> WMMA-B-fragment-native f16 layout:
//   Wsw[((ks*N + col)*32) + t]  <->  k = 32*ks + t; elem i reads t = 8*hi + i (i<8), 16 + 8*hi + i-8 (i>=8)
// so a wave's B fragment is two contiguous b128 loads.
// ---------------------------------------------------------------------------
__global__ void wswz_kernel(const float* __restrict__ W, _Float16* __restrict__ Wsw,
                            int K, int N, int total)
{
    int idx = blockIdx.x * blockDim.x + threadIdx.x;
    if (idx >= total) return;
    int t     = idx & 31;
    int colks = idx >> 5;
    int col   = colks % N;
    int ks    = colks / N;
    int k     = ks * 32 + t;
    Wsw[idx] = (k < K) ? (_Float16)W[(size_t)k * N + col] : (_Float16)0.0f;
}

// ---------------------------------------------------------------------------
// DIN attention: one wave per batch item. ceil(L/16) tiles of 16 positions.
// GEMM1: cross[16x64] @ aW1[64x64]  -> 8 wmma
// GEMM2: h1[16x64]    @ aW2[64x32]  -> 4 wmma (LDS transpose in between)
// GEMM3: h2[16x32]    @ aW3[32x1]   -> VALU dot + cross-lane reduce
// ---------------------------------------------------------------------------
__global__ void __launch_bounds__(256) attn_kernel(
    const int* __restrict__ feedid, const int* __restrict__ seq_ids,
    const int* __restrict__ seq_len,
    const float* __restrict__ emb_feedid, const float* __restrict__ emb_seq,
    const float* __restrict__ aW1, const float* __restrict__ ab1,
    const float* __restrict__ aW2, const float* __restrict__ ab2,
    const float* __restrict__ aW3, const float* __restrict__ ab3,
    float* __restrict__ att_out, int B, int L)
{
    __shared__ __align__(16) _Float16 h1s[8][16 * 64]; // per-wave h1 tile (2KB)
    __shared__ float scs[8][16];                        // per-wave scores

    const int lane = threadIdx.x & 31;
    const int wv   = threadIdx.x >> 5;
    const int b    = blockIdx.x * 8 + wv;
    const int c    = lane & 15;       // column / row-in-tile index
    const int hi   = lane >> 4;       // lane group (0 or 1)
    const int hi8  = hi * 8;

    // loop-invariant B fragments (lane = column, elem i<8 -> K = 8*hi + i, i>=8 -> K = 16 + 8*hi + i-8)
    v16h B1[2][4];
    #pragma unroll
    for (int kc = 0; kc < 2; ++kc)
        #pragma unroll
        for (int nt = 0; nt < 4; ++nt)
            #pragma unroll
            for (int i = 0; i < 16; ++i) {
                int k = kc * 32 + (i < 8 ? hi * 8 + i : 16 + hi * 8 + (i - 8));
                B1[kc][nt][i] = (_Float16)aW1[k * 64 + nt * 16 + c];
            }
    v16h B2[2][2];
    #pragma unroll
    for (int kc = 0; kc < 2; ++kc)
        #pragma unroll
        for (int nt = 0; nt < 2; ++nt)
            #pragma unroll
            for (int i = 0; i < 16; ++i) {
                int k = kc * 32 + (i < 8 ? hi * 8 + i : 16 + hi * 8 + (i - 8));
                B2[kc][nt][i] = (_Float16)aW2[k * 32 + nt * 16 + c];
            }
    const float w3a = aW3[c];
    const float w3b = aW3[16 + c];
    float b1v[4];
    #pragma unroll
    for (int nt = 0; nt < 4; ++nt) b1v[nt] = ab1[nt * 16 + c];
    float b2v[2];
    #pragma unroll
    for (int nt = 0; nt < 2; ++nt) b2v[nt] = ab2[nt * 16 + c];
    const float b3 = ab3[0];

    if (b >= B) return;

    const int slen = seq_len[b];
    const float4* qp = (const float4*)(emb_feedid + (size_t)feedid[b] * 16 + hi8);
    float4 q0 = qp[0], q1 = qp[1];
    float qf[8] = {q0.x, q0.y, q0.z, q0.w, q1.x, q1.y, q1.z, q1.w};
    _Float16 qh[8];
    #pragma unroll
    for (int i = 0; i < 8; ++i) qh[i] = (_Float16)qf[i];

    float attacc[8] = {0, 0, 0, 0, 0, 0, 0, 0};
    const int m = c;  // row of this lane within the 16-row tile

    for (int l0 = 0; l0 < L; l0 += 16) {
        int l   = l0 + m;
        int lcl = l < L ? l : L - 1;
        int sidx = seq_ids[(size_t)b * L + lcl];
        // warm next tile's index line (global_prefetch_b8)
        if (l0 + 16 < L)
            __builtin_prefetch(&seq_ids[(size_t)b * L + l0 + 16 + m], 0, 3);
        const float4* kp = (const float4*)(emb_seq + (size_t)sidx * 16 + hi8);
        float4 k0 = kp[0], k1 = kp[1];
        float kf[8] = {k0.x, k0.y, k0.z, k0.w, k1.x, k1.y, k1.z, k1.w};
        _Float16 kh[8];
        #pragma unroll
        for (int i = 0; i < 8; ++i) kh[i] = (_Float16)kf[i];

        // A fragments for cross = [q | key | q-key | q*key] (K = 64)
        v16h A0, A1;
        #pragma unroll
        for (int i = 0; i < 8; ++i) {
            A0[i]     = qh[i];
            A0[8 + i] = kh[i];
            A1[i]     = qh[i] - kh[i];
            A1[8 + i] = qh[i] * kh[i];
        }

        // GEMM1 + bias + relu
        v8f acc[4];
        #pragma unroll
        for (int nt = 0; nt < 4; ++nt) {
            v8f a = {};
            a = __builtin_amdgcn_wmma_f32_16x16x32_f16(false, A0, false, B1[0][nt],
                                                       (short)0, a, false, false);
            a = __builtin_amdgcn_wmma_f32_16x16x32_f16(false, A1, false, B1[1][nt],
                                                       (short)0, a, false, false);
            acc[nt] = a;
        }
        // transpose h1 through LDS: C-layout -> A-layout, f32 -> f16
        #pragma unroll
        for (int nt = 0; nt < 4; ++nt)
            #pragma unroll
            for (int r = 0; r < 8; ++r) {
                float hv = acc[nt][r] + b1v[nt];
                hv = hv > 0.0f ? hv : 0.0f;
                h1s[wv][(r + 8 * hi) * 64 + nt * 16 + c] = (_Float16)hv;
            }
        v16h A2[2];
        #pragma unroll
        for (int kc = 0; kc < 2; ++kc) {
            v8h lo = *(const v8h*)&h1s[wv][m * 64 + kc * 32 + hi8];
            v8h hh = *(const v8h*)&h1s[wv][m * 64 + kc * 32 + 16 + hi8];
            #pragma unroll
            for (int i = 0; i < 8; ++i) { A2[kc][i] = lo[i]; A2[kc][8 + i] = hh[i]; }
        }

        // GEMM2 + bias + relu, fused GEMM3 dot with aW3
        v8f acc2[2];
        #pragma unroll
        for (int nt = 0; nt < 2; ++nt) {
            v8f a = {};
            a = __builtin_amdgcn_wmma_f32_16x16x32_f16(false, A2[0], false, B2[0][nt],
                                                       (short)0, a, false, false);
            a = __builtin_amdgcn_wmma_f32_16x16x32_f16(false, A2[1], false, B2[1][nt],
                                                       (short)0, a, false, false);
            acc2[nt] = a;
        }
        float part[8];
        #pragma unroll
        for (int r = 0; r < 8; ++r) {
            float x0 = acc2[0][r] + b2v[0]; x0 = x0 > 0.0f ? x0 : 0.0f;
            float x1 = acc2[1][r] + b2v[1]; x1 = x1 > 0.0f ? x1 : 0.0f;
            part[r] = x0 * w3a + x1 * w3b;
        }
        #pragma unroll
        for (int r = 0; r < 8; ++r) {
            part[r] += __shfl_xor(part[r], 1, 32);
            part[r] += __shfl_xor(part[r], 2, 32);
            part[r] += __shfl_xor(part[r], 4, 32);
            part[r] += __shfl_xor(part[r], 8, 32);
        }
        // scatter scores (group hi holds rows 8*hi + r) into LDS scoreboard
        #pragma unroll
        for (int r = 0; r < 8; ++r)
            if (c == r) scs[wv][hi * 8 + r] = part[r] + b3;
        // masked weight for this lane's own row, accumulate weighted keys
        float wt = (l < slen) ? scs[wv][m] : 0.0f;
        #pragma unroll
        for (int i = 0; i < 8; ++i) attacc[i] += wt * kf[i];
    }

    #pragma unroll
    for (int i = 0; i < 8; ++i) {
        attacc[i] += __shfl_xor(attacc[i], 1, 32);
        attacc[i] += __shfl_xor(attacc[i], 2, 32);
        attacc[i] += __shfl_xor(attacc[i], 4, 32);
        attacc[i] += __shfl_xor(attacc[i], 8, 32);
    }
    if (c == 0) {
        #pragma unroll
        for (int i = 0; i < 8; ++i)
            att_out[(size_t)b * 16 + hi8 + i] = attacc[i];
    }
}

// ---------------------------------------------------------------------------
// Build net0 [B x 96] f16 (82 real cols + zero pad) and l2 regularizer partial
// ---------------------------------------------------------------------------
__global__ void __launch_bounds__(128) concat_kernel(
    const float* __restrict__ dense,
    const int* __restrict__ userid, const int* __restrict__ device,
    const int* __restrict__ authorid, const int* __restrict__ song,
    const int* __restrict__ singer, const int* __restrict__ tag,
    const int* __restrict__ feedid,
    const float* __restrict__ e_user, const float* __restrict__ e_dev,
    const float* __restrict__ e_auth, const float* __restrict__ e_song,
    const float* __restrict__ e_sing, const float* __restrict__ e_tag,
    const float* __restrict__ e_feed,
    const float* __restrict__ att_out,
    _Float16* __restrict__ net0, float* __restrict__ l2sum)
{
    __shared__ float red[128];
    const int row = blockIdx.x;
    const int c = threadIdx.x;
    float v = 0.0f;
    if (c < 16)       v = dense[(size_t)row * 16 + c];
    else if (c < 32)  v = e_user[(size_t)userid[row] * 16 + (c - 16)];
    else if (c < 34)  v = e_dev[(size_t)device[row] * 2 + (c - 32)];
    else if (c < 38)  v = e_auth[(size_t)authorid[row] * 4 + (c - 34)];
    else if (c < 42)  v = e_song[(size_t)song[row] * 4 + (c - 38)];
    else if (c < 46)  v = e_sing[(size_t)singer[row] * 4 + (c - 42)];
    else if (c < 50)  v = e_tag[(size_t)tag[row] * 4 + (c - 46)];
    else if (c < 66)  v = e_feed[(size_t)feedid[row] * 16 + (c - 50)];
    else if (c < 82)  v = att_out[(size_t)row * 16 + (c - 66)];
    if (c < 96) net0[(size_t)row * 96 + c] = (_Float16)v;
    red[c] = (c >= 16 && c < 82) ? v * v : 0.0f;
    __syncthreads();
    for (int s = 64; s > 0; s >>= 1) {
        if (c < s) red[c] += red[c + s];
        __syncthreads();
    }
    if (c == 0) atomicAdd(l2sum, sqrtf(red[0]));
}

// ---------------------------------------------------------------------------
// WMMA GEMM with fused batch-stats epilogue:
//   C[MxN] f32 = A[MxKp] f16 @ Wsw (pre-swizzled f16 B fragments) + bias[N]
//   accS[col] += sum_rows(C), accQ[col] += sum_rows(C^2)
// One wave per 16x16 output tile.
// ---------------------------------------------------------------------------
__global__ void __launch_bounds__(256) gemm_kernel(
    const _Float16* __restrict__ A, const _Float16* __restrict__ Wsw,
    const float* __restrict__ bias, float* __restrict__ C,
    float* __restrict__ accS, float* __restrict__ accQ,
    int M, int K, int N, int Kp)
{
    const int lane = threadIdx.x & 31;
    const int wave = (blockIdx.x * 256 + threadIdx.x) >> 5;
    const int ntiles = N >> 4;
    const int mt = wave / ntiles;
    const int nt = wave % ntiles;
    if (mt >= (M >> 4)) return;
    const int c    = lane & 15;
    const int hi   = lane >> 4;
    const int hi8  = hi * 8;
    const int hi16 = hi * 16;
    const int row = mt * 16 + c;
    const int col = nt * 16 + c;

    v8f acc = {};
    const int ksteps = (K + 31) >> 5;
    for (int ks = 0; ks < ksteps; ++ks) {
        // A fragment (16-bit A layout): two contiguous b128 loads
        const _Float16* ap = A + (size_t)row * Kp + ks * 32;
        v8h lo = *(const v8h*)(ap + hi8);
        v8h hh = *(const v8h*)(ap + 16 + hi8);
        if (ks + 1 < ksteps)
            __builtin_prefetch(ap + 32, 0, 3);
        v16h a;
        #pragma unroll
        for (int i = 0; i < 8; ++i) { a[i] = lo[i]; a[8 + i] = hh[i]; }
        // B fragment: two contiguous b128 loads from the swizzled layout
        const _Float16* bp = Wsw + ((size_t)ks * N + col) * 32 + hi8;
        v8h b0 = *(const v8h*)bp;
        v8h b1 = *(const v8h*)(bp + 16);
        v16h bf;
        #pragma unroll
        for (int i = 0; i < 8; ++i) { bf[i] = b0[i]; bf[8 + i] = b1[i]; }
        acc = __builtin_amdgcn_wmma_f32_16x16x32_f16(false, a, false, bf,
                                                     (short)0, acc, false, false);
    }
    const float bv = bias[col];
    float s = 0.0f, q = 0.0f;
    #pragma unroll
    for (int r = 0; r < 8; ++r) {
        float val = acc[r] + bv;
        C[(size_t)(mt * 16 + r + 8 * hi) * N + col] = val;
        s += val;
        q += val * val;
    }
    atomicAdd(&accS[col], s);
    atomicAdd(&accQ[col], q);
}

// ---------------------------------------------------------------------------
// Dice in-place with fused stats of the dice output:
//   d = alpha*(1-p)*x + p*x, p = sigmoid((x-m)*rsqrt(v+eps))
//   accS2[c] += sum(d), accQ2[c] += sum(d^2)
// Column-strided grid: consecutive threads = consecutive columns (coalesced).
// ---------------------------------------------------------------------------
__global__ void dice_kernel(float* __restrict__ Z,
                            const float* __restrict__ accS, const float* __restrict__ accQ,
                            const float* __restrict__ alpha,
                            float* __restrict__ accS2, float* __restrict__ accQ2,
                            int Brows, int N, int chunk, float invB)
{
    const int c = blockIdx.y * blockDim.x + threadIdx.x;
    if (c >= N) return;
    const float m  = accS[c] * invB;
    const float v  = accQ[c] * invB - m * m;
    const float rs = rsqrtf(v + BN_EPS);
    const float al = alpha[c];
    const int r0 = blockIdx.x * chunk;
    int r1 = r0 + chunk; if (r1 > Brows) r1 = Brows;
    float s = 0.0f, q = 0.0f;
    for (int r = r0; r < r1; ++r) {
        float x = Z[(size_t)r * N + c];
        float p = 1.0f / (1.0f + expf(-(x - m) * rs));
        float d = al * (1.0f - p) * x + p * x;
        Z[(size_t)r * N + c] = d;
        s += d;
        q += d * d;
    }
    atomicAdd(&accS2[c], s);
    atomicAdd(&accQ2[c], q);
}

// ---------------------------------------------------------------------------
// Affine BN of dice output -> f16 activations for next GEMM
// ---------------------------------------------------------------------------
__global__ void affine_kernel(const float* __restrict__ D,
                              const float* __restrict__ accS, const float* __restrict__ accQ,
                              const float* __restrict__ g, const float* __restrict__ beta,
                              _Float16* __restrict__ out, int N, int total, float invB)
{
    const int i = blockIdx.x * blockDim.x + threadIdx.x;
    if (i >= total) return;
    const int c = i % N;
    const float m = accS[c] * invB;
    const float v = accQ[c] * invB - m * m;
    const float x = D[i];
    out[i] = (_Float16)(g[c] * ((x - m) * rsqrtf(v + BN_EPS)) + beta[c]);
}

// ---------------------------------------------------------------------------
// Output head: logit = a2 @ Wout + bout, prob = sigmoid(logit); l2 finalize
// ---------------------------------------------------------------------------
__global__ void out_kernel(const _Float16* __restrict__ a2,
                           const float* __restrict__ Wout, const float* __restrict__ bout,
                           const float* __restrict__ l2sum, float* __restrict__ out, int B)
{
    const int row = blockIdx.x * blockDim.x + threadIdx.x;
    if (row < B) {
        float acc = bout[0];
        #pragma unroll 4
        for (int k = 0; k < 128; ++k)
            acc += (float)a2[(size_t)row * 128 + k] * Wout[k];
        out[row]     = 1.0f / (1.0f + expf(-acc));
        out[B + row] = acc;
    }
    if (row == 0) out[2 * B] = 0.2f * l2sum[0] / (float)B;
}

// ---------------------------------------------------------------------------
extern "C" void kernel_launch(void* const* d_in, const int* in_sizes, int n_in,
                              void* d_out, int out_size, void* d_ws, size_t ws_size,
                              hipStream_t stream)
{
    const float* dense    = (const float*)d_in[0];
    const int*   userid   = (const int*)d_in[1];
    const int*   device   = (const int*)d_in[2];
    const int*   authorid = (const int*)d_in[3];
    const int*   song     = (const int*)d_in[4];
    const int*   singer   = (const int*)d_in[5];
    const int*   tag      = (const int*)d_in[6];
    const int*   feedid   = (const int*)d_in[7];
    const int*   seq_ids  = (const int*)d_in[8];
    const int*   seq_len  = (const int*)d_in[9];
    const float* e_user = (const float*)d_in[10];
    const float* e_dev  = (const float*)d_in[11];
    const float* e_auth = (const float*)d_in[12];
    const float* e_song = (const float*)d_in[13];
    const float* e_sing = (const float*)d_in[14];
    const float* e_tag  = (const float*)d_in[15];
    const float* e_feed = (const float*)d_in[16];
    const float* e_seq  = (const float*)d_in[17];
    const float* aW1 = (const float*)d_in[18];
    const float* ab1 = (const float*)d_in[19];
    const float* aW2 = (const float*)d_in[20];
    const float* ab2 = (const float*)d_in[21];
    const float* aW3 = (const float*)d_in[22];
    const float* ab3 = (const float*)d_in[23];
    const float* Wl[3]    = {(const float*)d_in[24], (const float*)d_in[29], (const float*)d_in[34]};
    const float* bl[3]    = {(const float*)d_in[25], (const float*)d_in[30], (const float*)d_in[35]};
    const float* alphal[3]= {(const float*)d_in[26], (const float*)d_in[31], (const float*)d_in[36]};
    const float* gl[3]    = {(const float*)d_in[27], (const float*)d_in[32], (const float*)d_in[37]};
    const float* betal[3] = {(const float*)d_in[28], (const float*)d_in[33], (const float*)d_in[38]};
    const float* Wout = (const float*)d_in[39];
    const float* bout = (const float*)d_in[40];
    float* out = (float*)d_out;

    const int B = in_sizes[1];
    const int L = in_sizes[8] / B;
    const float invB = 1.0f / (float)B;

    const int Ks[3]  = {82, 512, 256};
    const int Kps[3] = {96, 512, 256};
    const int Ns[3]  = {512, 256, 128};

    // workspace layout
    char* ws = (char*)d_ws;
    float*      att  = (float*)ws;                              // B*16 f32
    _Float16*   net0 = (_Float16*)(ws + (size_t)B * 16 * 4);    // B*96 f16
    size_t off  = (size_t)B * 16 * 4 + (size_t)B * 96 * 2;
    off = (off + 255) & ~(size_t)255;
    float*      zbuf = (float*)(ws + off);                      // B*512 f32
    off += (size_t)B * 512 * 4;
    _Float16*   actA = (_Float16*)(ws + off);                   // B*512 f16
    off += (size_t)B * 512 * 2;
    _Float16*   actB = (_Float16*)(ws + off);                   // B*512 f16
    off += (size_t)B * 512 * 2;
    float*      stats = (float*)(ws + off);                     // 12 slots x 1024 + l2
    float*      l2sum = stats + 12 * 1024;
    off += (12 * 1024 + 16) * 4;
    off = (off + 255) & ~(size_t)255;
    _Float16*   Wsw[3];
    int         WswTot[3];
    for (int i = 0; i < 3; ++i) {
        int ksteps = (Ks[i] + 31) >> 5;
        WswTot[i] = ksteps * 32 * Ns[i];
        Wsw[i] = (_Float16*)(ws + off);
        off += (size_t)WswTot[i] * 2;
        off = (off + 255) & ~(size_t)255;
    }

    // 0) zero accumulators + swizzle weights (independent of data path)
    {
        int n = 12 * 1024 + 1;
        zero_kernel<<<(n + 255) / 256, 256, 0, stream>>>(stats, n);
    }
    for (int i = 0; i < 3; ++i)
        wswz_kernel<<<(WswTot[i] + 255) / 256, 256, 0, stream>>>(
            Wl[i], Wsw[i], Ks[i], Ns[i], WswTot[i]);

    // 1) attention
    attn_kernel<<<(B + 7) / 8, 256, 0, stream>>>(
        feedid, seq_ids, seq_len, e_feed, e_seq,
        aW1, ab1, aW2, ab2, aW3, ab3, att, B, L);
    // 2) concat + l2
    concat_kernel<<<B, 128, 0, stream>>>(
        dense, userid, device, authorid, song, singer, tag, feedid,
        e_user, e_dev, e_auth, e_song, e_sing, e_tag, e_feed,
        att, net0, l2sum);

    // 3) FCN stack: 82 -> 512 -> 256 -> 128 with Dice + affine BN
    const _Float16* gin[3]  = {net0, actA, actB};
    _Float16*       gout[3] = {actA, actB, actA};

    for (int layer = 0; layer < 3; ++layer) {
        const int K = Ks[layer], Kp = Kps[layer], N = Ns[layer];
        float* aS1 = stats + (layer * 4 + 0) * 1024;
        float* aQ1 = stats + (layer * 4 + 1) * 1024;
        float* aS2 = stats + (layer * 4 + 2) * 1024;
        float* aQ2 = stats + (layer * 4 + 3) * 1024;
        // GEMM -> z (f32) with fused stats of z
        {
            int tiles = (B / 16) * (N / 16);
            gemm_kernel<<<(tiles + 7) / 8, 256, 0, stream>>>(
                gin[layer], Wsw[layer], bl[layer], zbuf, aS1, aQ1, B, K, N, Kp);
        }
        // dice in-place with fused stats of dice output
        {
            const int chunk = 256;
            dim3 sgrid(B / chunk, (N + 255) / 256);
            dice_kernel<<<sgrid, 256, 0, stream>>>(
                zbuf, aS1, aQ1, alphal[layer], aS2, aQ2, B, N, chunk, invB);
        }
        // affine BN -> f16 activations
        {
            int total = B * N;
            affine_kernel<<<(total + 255) / 256, 256, 0, stream>>>(
                zbuf, aS2, aQ2, gl[layer], betal[layer], gout[layer], N, total, invB);
        }
    }

    // 4) output head + l2 finalize
    out_kernel<<<(B + 255) / 256, 256, 0, stream>>>(actA, Wout, bout, l2sum, out, B);
}
